// RNNLayer_21311627722984
// MI455X (gfx1250) — compile-verified, matrix operand order checked by reading, not run on hardware
//
#include <hip/hip_runtime.h>

// ---------------------------------------------------------------------------
// Elman RNN for MI455X (gfx1250, wave32, WMMA).
//   Phase 1: xh = x @ Wxh + bxh             (written into d_out)
//   Phase 2: h_t = relu(xh_t + h_{t-1}@Whh) persistent kernel, grid barrier
// All GEMMs via v_wmma_f32_16x16x32_bf16 (fp32 accumulate). Fragment loads
// are issued as arrays ahead of each WMMA block (wide s_clause + staggered
// s_wait_loadcnt, verified), k-loops are barrier-free, and f32->bf16
// conversion uses 2-wide vector truncation -> packed v_cvt_pk_bf16_f32.
// ---------------------------------------------------------------------------

typedef __attribute__((ext_vector_type(16))) __bf16       v16bf;
typedef __attribute__((ext_vector_type(2)))  __bf16       bf16x2;
typedef __attribute__((ext_vector_type(8)))  float        v8f;
typedef __attribute__((ext_vector_type(4)))  unsigned int uint4v;
typedef __attribute__((ext_vector_type(4)))  float        float4v;
typedef __attribute__((ext_vector_type(2)))  float        float2v;

union ABFrag { v16bf v; uint4v q[2]; bf16x2 p[8]; };  // 16 bf16 = 8 VGPRs
union CFrag  { v8f   v; float  f[8]; };               // 16x16 f32 accum

// One packed convert: fptrunc <2 x float> -> <2 x bfloat> = v_cvt_pk_bf16_f32
__device__ __forceinline__ bf16x2 cvt2(float lo, float hi) {
  float2v f; f.x = lo; f.y = hi;
  return __builtin_convertvector(f, bf16x2);
}

#define WMMA_BF16(A, B, C) \
  __builtin_amdgcn_wmma_f32_16x16x32_bf16(false, (A), false, (B), (short)0, (C), false, false)

// ---------------------------------------------------------------------------
// fp32 [K][N]  ->  bf16 [N][K]  (transpose + convert), 32x32 tiles via LDS.
// Transposed weights make the B fragment (lane = N column, K contiguous,
// mirroring the documented A layout) loadable with two b128 per lane.
// ---------------------------------------------------------------------------
__global__ void __launch_bounds__(256) cvt_transpose_kernel(
    const float* __restrict__ src, __bf16* __restrict__ dst, int K, int N) {
  __shared__ float tile[32][33];
  const int k0 = blockIdx.x * 32, n0 = blockIdx.y * 32;
  const int tx = threadIdx.x & 31, ty = threadIdx.x >> 5;  // 32 x 8
  for (int i = ty; i < 32; i += 8)
    tile[i][tx] = src[(size_t)(k0 + i) * N + (n0 + tx)];
  __syncthreads();
  for (int i = ty; i < 32; i += 8)
    dst[(size_t)(n0 + i) * K + (k0 + tx)] = (__bf16)tile[tx][i];
}

// ---------------------------------------------------------------------------
// Projection: out[BT,H] = x[BT,E] @ Wxh[E,H] + bxh.
// One WG (8 waves) per 16-row slab; each wave accumulates 8 16x16 tiles
// covering all 64 column tiles of H. Barrier-free: every wave loads the
// shared 16x32 fp32 x tile itself (WGP$/L2 absorb the 8x reuse) and converts
// with 8 packed cvt_pk; all 8 B fragments load before the 8 WMMAs.
// ---------------------------------------------------------------------------
__global__ void __launch_bounds__(256)
__attribute__((amdgpu_waves_per_eu(2)))
proj_kernel(
    const float*  __restrict__ x,      // [BT, E] fp32
    const __bf16* __restrict__ WxhT,   // [H, E]  bf16 (transposed)
    const float*  __restrict__ bxh,    // [H]
    float*        __restrict__ out) {  // [BT, H]
  constexpr int E = 512, H = 1024;

  const int tid  = threadIdx.x;
  const int lane = tid & 31, wave = tid >> 5;
  const size_t row0 = (size_t)blockIdx.x * 16;
  const int khalf = (lane >> 4) * 8;   // K offset for this lane half
  const int nlane = lane & 15;         // N / M position within tile

  CFrag acc[8];
#pragma unroll
  for (int j = 0; j < 8; ++j)
#pragma unroll
    for (int r = 0; r < 8; ++r) acc[j].f[r] = 0.0f;

  const float* arow = x + (row0 + nlane) * E + khalf;  // this lane's A row

  for (int ks = 0; ks < E; ks += 32) {
    // All 8 B fragments first (16 x global_load_b128 -> wide clause).
    ABFrag b[8];
#pragma unroll
    for (int j = 0; j < 8; ++j) {
      const __bf16* bp =
          WxhT + (size_t)((j * 8 + wave) * 16 + nlane) * E + ks + khalf;
      b[j].q[0] = *(const uint4v*)bp;
      b[j].q[1] = *(const uint4v*)(bp + 16);
    }

    // A fragment: 16 fp32 loads (4 x b128) -> 8 packed bf16 converts.
    const float4v f0 = *(const float4v*)(arow + ks);
    const float4v f1 = *(const float4v*)(arow + ks + 4);
    const float4v f2 = *(const float4v*)(arow + ks + 16);
    const float4v f3 = *(const float4v*)(arow + ks + 20);
    ABFrag a;
    a.p[0] = cvt2(f0.x, f0.y); a.p[1] = cvt2(f0.z, f0.w);
    a.p[2] = cvt2(f1.x, f1.y); a.p[3] = cvt2(f1.z, f1.w);
    a.p[4] = cvt2(f2.x, f2.y); a.p[5] = cvt2(f2.z, f2.w);
    a.p[6] = cvt2(f3.x, f3.y); a.p[7] = cvt2(f3.z, f3.w);

    // 8 WMMAs with staggered waits against the B clause above.
#pragma unroll
    for (int j = 0; j < 8; ++j)
      acc[j].v = WMMA_BF16(a.v, b[j].v, acc[j].v);
  }

  const int rowoff = (lane >> 4) * 8;  // C layout: VGPR r -> M = r + 8*half
#pragma unroll
  for (int j = 0; j < 8; ++j) {
    const int colj = (j * 8 + wave) * 16 + nlane;
    const float bias = bxh[colj];
#pragma unroll
    for (int r = 0; r < 8; ++r)
      out[(row0 + r + rowoff) * (size_t)H + colj] = acc[j].f[r] + bias;
  }
}

// ---------------------------------------------------------------------------
// Persistent recurrent scan. 64 WGs; WG w owns output columns [16w,16w+16).
// Its Whh^T column slab (16 x 1024 bf16 = 32 KB) stays in LDS for all 512
// steps. Each of 8 waves owns two 16x16 M-tiles (covers B=256 rows).
// K is processed in 128-wide chunks: 12 fragments (4 LDS B + 8 global A)
// load as arrays before 8 interleaved WMMAs. xh_t is loaded at the top of
// each step and added only in the epilogue (hidden behind 64 WMMAs).
// ---------------------------------------------------------------------------
__global__ void __launch_bounds__(256)
__attribute__((amdgpu_waves_per_eu(2)))
rnn_scan_kernel(
    float*        __restrict__ out,    // [B,T,H]: holds xh, overwritten with h
    const __bf16* __restrict__ WhhT,   // [H, H] bf16 (transposed)
    __bf16*       __restrict__ hbuf0,  // [B, H] (zero at t=0)
    __bf16*       __restrict__ hbuf1,  // [B, H]
    unsigned*     __restrict__ counter) {
  constexpr int T = 512, H = 1024;
  __shared__ __attribute__((aligned(16))) __bf16 sWT[16 * H];  // 32 KB

  const int tid  = threadIdx.x;
  const int lane = tid & 31, wave = tid >> 5;
  const int n0   = blockIdx.x * 16;
  const unsigned nwg = gridDim.x;

  // Stage this WG's Whh^T slab into LDS (2048 x b128), one-time cost.
  for (int i = tid; i < (16 * H) / 8; i += 256) {
    const int c = i >> 7, q = i & 127;
    ((uint4v*)sWT)[i] = ((const uint4v*)(WhhT + (size_t)(n0 + c) * H))[q];
  }
  __syncthreads();

  const int nlane  = lane & 15;
  const int khalf  = (lane >> 4) * 8;
  const int rowoff = (lane >> 4) * 8;
  const int col    = n0 + nlane;
  const int mb0    = (wave * 2) * 16;
  const int mb1    = (wave * 2 + 1) * 16;
  const __bf16* sWrow = sWT + nlane * H + khalf;  // this lane's B base in LDS

  for (int t = 0; t < T; ++t) {
    const __bf16* __restrict__ hp = (t & 1) ? hbuf1 : hbuf0;
    __bf16*       __restrict__ hn = (t & 1) ? hbuf0 : hbuf1;

    // Issue xh_t loads now; consumed only after the k-loop.
    CFrag xh0, xh1;
#pragma unroll
    for (int r = 0; r < 8; ++r) {
      xh0.f[r] = out[((size_t)(mb0 + r + rowoff) * T + t) * H + col];
      xh1.f[r] = out[((size_t)(mb1 + r + rowoff) * T + t) * H + col];
    }

    CFrag c0, c1;
#pragma unroll
    for (int r = 0; r < 8; ++r) { c0.f[r] = 0.0f; c1.f[r] = 0.0f; }

    const __bf16* ar0 = hp + (size_t)(mb0 + nlane) * H + khalf;
    const __bf16* ar1 = hp + (size_t)(mb1 + nlane) * H + khalf;

    // 128-wide K chunks: batch-load 4 B (ds) + 8 A (global) fragments, then
    // 8 WMMAs interleaved over the two accumulators.
    for (int kb = 0; kb < H; kb += 128) {
      ABFrag b[4], a0[4], a1[4];
#pragma unroll
      for (int s = 0; s < 4; ++s) {
        const int k = kb + s * 32;
        b[s].q[0]  = *(const uint4v*)(sWrow + k);
        b[s].q[1]  = *(const uint4v*)(sWrow + k + 16);
        a0[s].q[0] = *(const uint4v*)(ar0 + k);
        a0[s].q[1] = *(const uint4v*)(ar0 + k + 16);
        a1[s].q[0] = *(const uint4v*)(ar1 + k);
        a1[s].q[1] = *(const uint4v*)(ar1 + k + 16);
      }
#pragma unroll
      for (int s = 0; s < 4; ++s) {
        c0.v = WMMA_BF16(a0[s].v, b[s].v, c0.v);
        c1.v = WMMA_BF16(a1[s].v, b[s].v, c1.v);
      }
    }

    // ReLU(xh + h@Whh); write h_t to d_out (fp32) and next h buffer (bf16).
#pragma unroll
    for (int r = 0; r < 8; ++r) {
      const float v0 = fmaxf(c0.f[r] + xh0.f[r], 0.0f);
      const float v1 = fmaxf(c1.f[r] + xh1.f[r], 0.0f);
      out[((size_t)(mb0 + r + rowoff) * T + t) * H + col] = v0;
      out[((size_t)(mb1 + r + rowoff) * T + t) * H + col] = v1;
      hn[(size_t)(mb0 + r + rowoff) * H + col] = (__bf16)v0;
      hn[(size_t)(mb1 + r + rowoff) * H + col] = (__bf16)v1;
    }

    // ---- grid-wide barrier between timesteps (monotonic counter) ----
    __builtin_amdgcn_fence(__ATOMIC_RELEASE, "agent");  // flush h stores
    __syncthreads();
    if (tid == 0) {
      __hip_atomic_fetch_add(counter, 1u, __ATOMIC_RELAXED,
                             __HIP_MEMORY_SCOPE_AGENT);
      const unsigned target = (unsigned)(t + 1) * nwg;
      while (__hip_atomic_load(counter, __ATOMIC_RELAXED,
                               __HIP_MEMORY_SCOPE_AGENT) < target) {
        __builtin_amdgcn_s_sleep(2);
      }
    }
    __syncthreads();
    __builtin_amdgcn_fence(__ATOMIC_ACQUIRE, "agent");  // invalidate for h loads
  }
}

// ---------------------------------------------------------------------------
extern "C" void kernel_launch(void* const* d_in, const int* in_sizes, int n_in,
                              void* d_out, int out_size, void* d_ws, size_t ws_size,
                              hipStream_t stream) {
  (void)in_sizes; (void)n_in; (void)out_size; (void)ws_size;
  const float* x   = (const float*)d_in[0];  // [256,512,512]
  const float* Wxh = (const float*)d_in[1];  // [512,1024]
  const float* bxh = (const float*)d_in[2];  // [1024]
  const float* Whh = (const float*)d_in[3];  // [1024,1024]
  float* out = (float*)d_out;                // [256,512,1024]

  constexpr int B = 256, T = 512, E = 512, H = 1024;

  // Workspace layout (~4 MB): counter | WxhT bf16 | WhhT bf16 | h0 | h1
  char* ws = (char*)d_ws;
  unsigned* counter = (unsigned*)ws;
  __bf16* WxhT = (__bf16*)(ws + 256);
  __bf16* WhhT = (__bf16*)(ws + 256 + (size_t)E * H * 2);
  __bf16* h0   = (__bf16*)(ws + 256 + (size_t)E * H * 2 + (size_t)H * H * 2);
  __bf16* h1   = h0 + (size_t)B * H;

  hipMemsetAsync(counter, 0, 256, stream);
  hipMemsetAsync(h0, 0, (size_t)B * H * sizeof(__bf16), stream);  // h_{-1} = 0

  cvt_transpose_kernel<<<dim3(E / 32, H / 32), 256, 0, stream>>>(Wxh, WxhT, E, H);
  cvt_transpose_kernel<<<dim3(H / 32, H / 32), 256, 0, stream>>>(Whh, WhhT, H, H);
  proj_kernel<<<(B * T) / 16, 256, 0, stream>>>(x, WxhT, bxh, out);
  rnn_scan_kernel<<<H / 16, 256, 0, stream>>>(out, WhhT, h0, h1, counter);
}
